// FraudDetectionHybrid_65481071400088
// MI455X (gfx1250) — matrix operand order repaired and checked by reading, not compiled, not run on hardware
//
#include <hip/hip_runtime.h>
#include <math.h>

// ---------------------------------------------------------------------------
// FraudDetectionHybrid on MI455X (gfx1250): 2-layer tanh MLP + RBF gram + head
//   h1 = tanh(x @ W1^T + b1)        [4096 x 4096]
//   h2 = tanh(h1 @ W2^T + b2)       [4096 x 4096]
//   out[i] = sum_j exp(-g*(|h2_i|^2 + |h2_j|^2 - 2 h2_i.h2_j)) * Wc[j] + bc
//
// f16 storage + v_wmma_f32_16x16x32_f16 (f32 accumulate), 2x4 register
// blocking (341 FLOP per byte from L2), manual double-buffering, and —
// key change this round — ONLY TWO loop-carried address registers per
// kernel: all 12 fragment loads per k-step are base + compile-time
// immediate offsets (tile stride 16*K*2 B and k-advance fold into the
// 24-bit ioffset), so the k-loop has no per-stream 64-bit address VALU,
// no pointer spills, and no WMMA->VALU hazard NOPs.
// ---------------------------------------------------------------------------

typedef __attribute__((ext_vector_type(16))) _Float16 v16h;
typedef __attribute__((ext_vector_type(8)))  _Float16 v8h;
typedef __attribute__((ext_vector_type(8)))  float    v8f;

#define N_ROWS 4096
#define IN_DIM 512
#define HID    4096
#define GAMMA  5.0e-4f

#define WMMA_F16(a, b, c) \
    __builtin_amdgcn_wmma_f32_16x16x32_f16(false, (a), false, (b), (short)0, (c), false, false)

// --- WMMA fragment loads (wave32; this file's contiguous k order, same for A and B) -------------
// A (16x32, MxK): lane l -> row l&15, k-base (l>>4)*16;
//   halves 0..15 = k+0..15 (contiguous)            -> 16B loads at off, off+16B.
// B (32x16, KxN) from W stored [N,K]: lane l -> col l&15, k-base (l>>4)*16;
//   halves 0..15 = k+0..15 (contiguous)            -> 16B loads at off, off+16B.

__device__ __forceinline__ v16h frag_a_at(const _Float16* __restrict__ p, size_t off) {
    v8h lo = *reinterpret_cast<const v8h*>(p + off);
    v8h hi = *reinterpret_cast<const v8h*>(p + off + 8);
    v16h a;
#pragma unroll
    for (int i = 0; i < 8; ++i) { a[i] = lo[i]; a[8 + i] = hi[i]; }
    return a;
}
__device__ __forceinline__ v16h frag_b_at(const _Float16* __restrict__ p, size_t off) {
    v8h lo = *reinterpret_cast<const v8h*>(p + off);
    v8h hi = *reinterpret_cast<const v8h*>(p + off + 8);
    v16h b;
#pragma unroll
    for (int i = 0; i < 8; ++i) { b[i] = lo[i]; b[8 + i] = hi[i]; }
    return b;
}

// --- elementwise f32 -> f16 -------------------------------------------------
__global__ void fdh_cvt_f16(const float* __restrict__ src,
                            _Float16* __restrict__ dst, int n) {
    for (int i = blockIdx.x * blockDim.x + threadIdx.x; i < n;
         i += gridDim.x * blockDim.x)
        dst[i] = (_Float16)src[i];
}

// --- GEMM: C[4096x4096] = tanh(A @ W^T + bias), K compile-time -------------
// Each wave: 32x64 output (2 M-tiles x 4 N-tiles), double-buffered k-loop,
// 2 loop-carried pointers, all fragment loads at immediate offsets.
// grid: (4096/64, 4096/128), block 128 (4 waves; wave picks the M strip)
template <int K>
__global__ __launch_bounds__(128)
void fdh_gemm_bias_tanh(const _Float16* __restrict__ A,
                        const _Float16* __restrict__ W,
                        const float* __restrict__ bias,
                        _Float16* __restrict__ C) {
    const int lane = threadIdx.x & 31;
    const int wave = threadIdx.x >> 5;
    const int nb   = blockIdx.x * 4;
    const int mb   = (blockIdx.y * 4 + wave) * 2;
    constexpr size_t TS = (size_t)16 * K;   // halves between 16-row tiles

    const _Float16* pa = A + (size_t)(mb * 16 + (lane & 15)) * K + ((lane >> 4) << 4);
    const _Float16* pb = W + (size_t)(nb * 16 + (lane & 15)) * K + ((lane >> 4) << 4);

    v8f acc[2][4] = {};
    v16h a0 = frag_a_at(pa, 0 * TS), a1 = frag_a_at(pa, 1 * TS);
    v16h b0 = frag_b_at(pb, 0 * TS), b1 = frag_b_at(pb, 1 * TS);
    v16h b2 = frag_b_at(pb, 2 * TS), b3 = frag_b_at(pb, 3 * TS);

    for (int it = 0; it < K / 32 - 1; ++it) {
        // next k-step: 6 loads, all base+imm (k-advance is the +32)
        v16h na0 = frag_a_at(pa, 0 * TS + 32), na1 = frag_a_at(pa, 1 * TS + 32);
        v16h nb0 = frag_b_at(pb, 0 * TS + 32), nb1 = frag_b_at(pb, 1 * TS + 32);
        v16h nb2 = frag_b_at(pb, 2 * TS + 32), nb3 = frag_b_at(pb, 3 * TS + 32);

        acc[0][0] = WMMA_F16(a0, b0, acc[0][0]);
        acc[1][0] = WMMA_F16(a1, b0, acc[1][0]);
        acc[0][1] = WMMA_F16(a0, b1, acc[0][1]);
        acc[1][1] = WMMA_F16(a1, b1, acc[1][1]);
        acc[0][2] = WMMA_F16(a0, b2, acc[0][2]);
        acc[1][2] = WMMA_F16(a1, b2, acc[1][2]);
        acc[0][3] = WMMA_F16(a0, b3, acc[0][3]);
        acc[1][3] = WMMA_F16(a1, b3, acc[1][3]);

        pa += 32; pb += 32;
        a0 = na0; a1 = na1; b0 = nb0; b1 = nb1; b2 = nb2; b3 = nb3;
    }
    acc[0][0] = WMMA_F16(a0, b0, acc[0][0]);
    acc[1][0] = WMMA_F16(a1, b0, acc[1][0]);
    acc[0][1] = WMMA_F16(a0, b1, acc[0][1]);
    acc[1][1] = WMMA_F16(a1, b1, acc[1][1]);
    acc[0][2] = WMMA_F16(a0, b2, acc[0][2]);
    acc[1][2] = WMMA_F16(a1, b2, acc[1][2]);
    acc[0][3] = WMMA_F16(a0, b3, acc[0][3]);
    acc[1][3] = WMMA_F16(a1, b3, acc[1][3]);

    const int n = lane & 15;
#pragma unroll
    for (int j = 0; j < 4; ++j) {
        const int   col = (nb + j) * 16 + n;
        const float bv  = bias[col];
#pragma unroll
        for (int im = 0; im < 2; ++im) {
            const int mbase = (mb + im) * 16 + ((lane >> 4) << 3);
#pragma unroll
            for (int r = 0; r < 8; ++r) {
                float v = tanhf(acc[im][j][r] + bv);
                C[(size_t)(mbase + r) * HID + col] = (_Float16)v;
            }
        }
    }
}

// --- row squared norms of h2 (f16 in, f32 out); one wave per row -----------
__global__ void fdh_row_sqnorm(const _Float16* __restrict__ H,
                               float* __restrict__ norms) {
    const int row  = blockIdx.x;
    const int lane = threadIdx.x;
    const _Float16* p = H + (size_t)row * HID;
    float s = 0.0f;
    for (int c = lane; c < HID / 8; c += 32) {
        v8h v = *reinterpret_cast<const v8h*>(p + c * 8);
#pragma unroll
        for (int i = 0; i < 8; ++i) { float f = (float)v[i]; s += f * f; }
    }
#pragma unroll
    for (int m = 16; m >= 1; m >>= 1) s += __shfl_xor(s, m, 32);
    if (lane == 0) norms[row] = s;
}

__global__ void fdh_init_out(float* __restrict__ out,
                             const float* __restrict__ bc, int n) {
    int i = blockIdx.x * blockDim.x + threadIdx.x;
    if (i < n) out[i] = bc[0];
}

// --- Fused gram + classifier ------------------------------------------------
// Each wave: 2 i-tiles x 4 j-tiles (32x64 of the gram), double-buffered,
// 2 loop-carried pointers. RBF+Wc applied in registers; lane reduction over
// the 16 columns, then atomicAdd into out.
// grid: (64, 32), block 128: blockIdx.x -> j-chunk, blockIdx.y*4+wave -> i-strip
__global__ __launch_bounds__(128)
void fdh_gram_classify(const _Float16* __restrict__ H,
                       const float* __restrict__ norms,
                       const float* __restrict__ Wc,
                       float* __restrict__ out) {
    const int lane = threadIdx.x & 31;
    const int wave = threadIdx.x >> 5;
    const int ib   = (blockIdx.y * 4 + wave) * 2;   // first of 2 i-tiles
    const int jb   = blockIdx.x * 4;                // 4 j-tiles per wave
    const int rsel = (lane >> 4) << 3;
    constexpr size_t TS = (size_t)16 * HID;

    const _Float16* pa = H + (size_t)(ib * 16 + (lane & 15)) * HID + ((lane >> 4) << 4);
    const _Float16* pb = H + (size_t)(jb * 16 + (lane & 15)) * HID + ((lane >> 4) << 4);

    v8f S[2][4] = {};
    v16h a0 = frag_a_at(pa, 0 * TS), a1 = frag_a_at(pa, 1 * TS);
    v16h b0 = frag_b_at(pb, 0 * TS), b1 = frag_b_at(pb, 1 * TS);
    v16h b2 = frag_b_at(pb, 2 * TS), b3 = frag_b_at(pb, 3 * TS);

    for (int it = 0; it < HID / 32 - 1; ++it) {
        v16h na0 = frag_a_at(pa, 0 * TS + 32), na1 = frag_a_at(pa, 1 * TS + 32);
        v16h nb0 = frag_b_at(pb, 0 * TS + 32), nb1 = frag_b_at(pb, 1 * TS + 32);
        v16h nb2 = frag_b_at(pb, 2 * TS + 32), nb3 = frag_b_at(pb, 3 * TS + 32);

        S[0][0] = WMMA_F16(a0, b0, S[0][0]);
        S[1][0] = WMMA_F16(a1, b0, S[1][0]);
        S[0][1] = WMMA_F16(a0, b1, S[0][1]);
        S[1][1] = WMMA_F16(a1, b1, S[1][1]);
        S[0][2] = WMMA_F16(a0, b2, S[0][2]);
        S[1][2] = WMMA_F16(a1, b2, S[1][2]);
        S[0][3] = WMMA_F16(a0, b3, S[0][3]);
        S[1][3] = WMMA_F16(a1, b3, S[1][3]);

        pa += 32; pb += 32;
        a0 = na0; a1 = na1; b0 = nb0; b1 = nb1; b2 = nb2; b3 = nb3;
    }
    S[0][0] = WMMA_F16(a0, b0, S[0][0]);
    S[1][0] = WMMA_F16(a1, b0, S[1][0]);
    S[0][1] = WMMA_F16(a0, b1, S[0][1]);
    S[1][1] = WMMA_F16(a1, b1, S[1][1]);
    S[0][2] = WMMA_F16(a0, b2, S[0][2]);
    S[1][2] = WMMA_F16(a1, b2, S[1][2]);
    S[0][3] = WMMA_F16(a0, b3, S[0][3]);
    S[1][3] = WMMA_F16(a1, b3, S[1][3]);

    float ni[2][8];
#pragma unroll
    for (int im = 0; im < 2; ++im)
#pragma unroll
        for (int r = 0; r < 8; ++r)
            ni[im][r] = norms[(ib + im) * 16 + rsel + r];

    const int n = lane & 15;
    float acc[2][8] = {};
#pragma unroll
    for (int t = 0; t < 4; ++t) {
        const int   col = (jb + t) * 16 + n;
        const float wj  = Wc[col];
        const float nj  = norms[col];
#pragma unroll
        for (int im = 0; im < 2; ++im)
#pragma unroll
            for (int r = 0; r < 8; ++r) {
                float sq = ni[im][r] + nj - 2.0f * S[im][t][r];
                acc[im][r] += __expf(-GAMMA * sq) * wj;
            }
    }

    // reduce over the 16 columns held by each half-wave, then atomically add
#pragma unroll
    for (int im = 0; im < 2; ++im)
#pragma unroll
        for (int r = 0; r < 8; ++r) {
            float v = acc[im][r];
            v += __shfl_xor(v, 1, 32);
            v += __shfl_xor(v, 2, 32);
            v += __shfl_xor(v, 4, 32);
            v += __shfl_xor(v, 8, 32);
            if ((lane & 15) == 0)
                atomicAdd(&out[(ib + im) * 16 + rsel + r], v);
        }
}

// ---------------------------------------------------------------------------
extern "C" void kernel_launch(void* const* d_in, const int* in_sizes, int n_in,
                              void* d_out, int out_size, void* d_ws, size_t ws_size,
                              hipStream_t stream) {
    const float* x  = (const float*)d_in[0];
    const float* W1 = (const float*)d_in[1];
    const float* b1 = (const float*)d_in[2];
    const float* W2 = (const float*)d_in[3];
    const float* b2 = (const float*)d_in[4];
    const float* Wc = (const float*)d_in[5];
    const float* bc = (const float*)d_in[6];
    float* out = (float*)d_out;

    char* ws = (char*)d_ws;
    _Float16* xh  = (_Float16*)(ws + ((size_t)0   << 20)); //  4 MiB
    _Float16* w1h = (_Float16*)(ws + ((size_t)4   << 20)); //  4 MiB
    _Float16* h1h = (_Float16*)(ws + ((size_t)8   << 20)); // 32 MiB
    _Float16* w2h = (_Float16*)(ws + ((size_t)40  << 20)); // 32 MiB
    _Float16* h2h = (_Float16*)(ws + ((size_t)72  << 20)); // 32 MiB
    float*    nrm = (float*)   (ws + ((size_t)104 << 20)); // 16 KiB

    fdh_cvt_f16<<<2048, 256, 0, stream>>>(x,  xh,  N_ROWS * IN_DIM);
    fdh_cvt_f16<<<2048, 256, 0, stream>>>(W1, w1h, HID * IN_DIM);
    fdh_cvt_f16<<<4096, 256, 0, stream>>>(W2, w2h, HID * HID);

    // h1 = tanh(x @ W1^T + b1) : M=N=4096, K=512
    fdh_gemm_bias_tanh<IN_DIM><<<dim3(HID / 64, N_ROWS / 128), 128, 0, stream>>>(
        xh, w1h, b1, h1h);

    // h2 = tanh(h1 @ W2^T + b2) : M=N=K=4096
    fdh_gemm_bias_tanh<HID><<<dim3(HID / 64, N_ROWS / 128), 128, 0, stream>>>(
        h1h, w2h, b2, h2h);

    fdh_row_sqnorm<<<N_ROWS, 32, 0, stream>>>(h2h, nrm);
    fdh_init_out<<<(N_ROWS + 255) / 256, 256, 0, stream>>>(out, bc, N_ROWS);

    // fused gram + classifier: 128 i-strips x 64 j-chunks
    fdh_gram_classify<<<dim3(64, 32), 128, 0, stream>>>(h2h, nrm, Wc, out);
}
